// GPTLanguageModel_73229192396843
// MI455X (gfx1250) — matrix operand order repaired and checked by reading, not compiled, not run on hardware
//
#include <hip/hip_runtime.h>
#include <hip/hip_bf16.h>

typedef __bf16 bf16_t;
typedef __attribute__((ext_vector_type(16))) __bf16 v16bf;
typedef __attribute__((ext_vector_type(8)))  float  v8f;

#define NEG_BIG (-3.0e38f)

__device__ __forceinline__ v8f v8f_zero() {
    v8f z;
#pragma unroll
    for (int i = 0; i < 8; ++i) z[i] = 0.0f;
    return z;
}

// ---------------------------------------------------------------------------
// f32 -> bf16 conversion (grid-stride)
// ---------------------------------------------------------------------------
__global__ __launch_bounds__(256)
void cvt_bf16_kernel(const float* __restrict__ in, bf16_t* __restrict__ out, size_t n) {
    size_t i = (size_t)blockIdx.x * blockDim.x + threadIdx.x;
    size_t stride = (size_t)gridDim.x * blockDim.x;
    for (; i < n; i += stride) out[i] = (bf16_t)in[i];
}

// ---------------------------------------------------------------------------
// x[b,t,:] = tok_table[idx[b,t],:] + pos_table[t,:]
// ---------------------------------------------------------------------------
__global__ __launch_bounds__(256)
void embed_kernel(const int* __restrict__ idx, const float* __restrict__ tok,
                  const float* __restrict__ pos, float* __restrict__ x,
                  int T, int D) {
    int row = blockIdx.x;                 // b*T + t
    int t = row % T;
    int token = idx[row];
    const float* tr = tok + (size_t)token * D;
    const float* pr = pos + (size_t)t * D;
    float* xr = x + (size_t)row * D;
    for (int c = threadIdx.x; c < D; c += 256) xr[c] = tr[c] + pr[c];
}

// ---------------------------------------------------------------------------
// LayerNorm (f32 stats) -> bf16 output with row stride; optionally append
// book vector in columns [D, D+Db).
// ---------------------------------------------------------------------------
__global__ __launch_bounds__(256)
void layernorm_kernel(const float* __restrict__ x, const float* __restrict__ gamma,
                      const float* __restrict__ beta, bf16_t* __restrict__ out,
                      int D, int outStride,
                      const float* __restrict__ book, const int* __restrict__ ibx,
                      int Db, int T) {
    int row = blockIdx.x;
    int tid = threadIdx.x;
    const float* xr = x + (size_t)row * D;
    __shared__ float red[256];

    float s = 0.0f;
    for (int c = tid; c < D; c += 256) s += xr[c];
    red[tid] = s; __syncthreads();
    for (int st = 128; st > 0; st >>= 1) {
        if (tid < st) red[tid] += red[tid + st];
        __syncthreads();
    }
    float mu = red[0] / (float)D;
    __syncthreads();

    float vs = 0.0f;
    for (int c = tid; c < D; c += 256) { float d0 = xr[c] - mu; vs += d0 * d0; }
    red[tid] = vs; __syncthreads();
    for (int st = 128; st > 0; st >>= 1) {
        if (tid < st) red[tid] += red[tid + st];
        __syncthreads();
    }
    float rstd = rsqrtf(red[0] / (float)D + 1e-5f);

    bf16_t* orow = out + (size_t)row * outStride;
    for (int c = tid; c < D; c += 256)
        orow[c] = (bf16_t)((xr[c] - mu) * rstd * gamma[c] + beta[c]);

    if (book != nullptr) {
        int bidx = ibx[row / T];
        const float* bv = book + (size_t)bidx * Db;
        for (int c = tid; c < Db; c += 256) orow[D + c] = (bf16_t)bv[c];
    }
}

// ---------------------------------------------------------------------------
// Tiled bf16 WMMA GEMM: C[M,N] = A[M,K] @ B[K,N] (+bias) (+residual) (ReLU)
// 128x128 tile per 256-thread block (8 waves); each wave owns 64x32.
// LDS layout is fragment-contiguous: A row-major [m][k], B transposed [n][k].
// Rows padded 32->40 elems (80B, 16B aligned): 20*n mod 64 banks => conflict-
// free b128 fragment loads across 16 lanes. Global tile loads are b128
// (all strides provably 16B aligned); next K-tile is prefetched
// (global_prefetch_b8) while the current one computes.
// ---------------------------------------------------------------------------
#define GT_M 128
#define GT_N 128
#define GT_K 32
#define GT_KP 40

__global__ __launch_bounds__(256)
void gemm_bf16_kernel(const bf16_t* __restrict__ A, const bf16_t* __restrict__ Bw,
                      const float* __restrict__ bias, const float* __restrict__ resid,
                      float* __restrict__ Cf, bf16_t* __restrict__ Cb,
                      int M, int N, int K, int relu) {
    __shared__ bf16_t As [GT_M][GT_KP];   // 10 KB, As[m][k]
    __shared__ bf16_t Bst[GT_N][GT_KP];   // 10 KB, Bst[n][k]  (B transposed)

    const int tid  = threadIdx.x;
    const int lane = tid & 31;
    const int wave = tid >> 5;
    const int wm   = wave & 1;   // 0..1 : 64-row half
    const int wn   = wave >> 1;  // 0..3 : 32-col strip
    const int g    = lane >> 4;  // lane-half
    const int l16  = lane & 15;
    const int tile_m = blockIdx.y * GT_M;
    const int tile_n = blockIdx.x * GT_N;

    v8f acc[4][2];
#pragma unroll
    for (int i = 0; i < 4; ++i)
#pragma unroll
        for (int j = 0; j < 2; ++j) acc[i][j] = v8f_zero();

    for (int k0 = 0; k0 < K; k0 += GT_K) {
        // A tile: row-major b128 copy (2 chunks of 8 bf16 per thread)
        for (int i = tid; i < GT_M * GT_K / 8; i += 256) {
            int r = (i * 8) / GT_K, c = (i * 8) % GT_K;
            *(uint4*)&As[r][c] =
                *(const uint4*)&A[(size_t)(tile_m + r) * K + k0 + c];
        }
        // B tile: b128 global read (8 consecutive n), transposed LDS scatter
        for (int i = tid; i < GT_K * GT_N / 8; i += 256) {
            int r = (i * 8) / GT_N, c = (i * 8) % GT_N;   // r = k, c = n
            uint4 w = *(const uint4*)&Bw[(size_t)(k0 + r) * N + tile_n + c];
            const bf16_t* e = (const bf16_t*)&w;
#pragma unroll
            for (int j = 0; j < 8; ++j) Bst[c + j][r] = e[j];
        }
        // prefetch next K-tile while this one computes
        if (k0 + GT_K < K) {
            int ra = (tid * 8) / GT_K, ca = (tid * 8) % GT_K;
            __builtin_prefetch(&A[(size_t)(tile_m + ra) * K + k0 + GT_K + ca], 0, 3);
            int rb = (tid * 8) / GT_N, cb = (tid * 8) % GT_N;
            __builtin_prefetch(&Bw[(size_t)(k0 + GT_K + rb) * N + tile_n + cb], 0, 3);
        }
        __syncthreads();

        v16bf af[4];
#pragma unroll
        for (int mi = 0; mi < 4; ++mi) {
            int m = wm * 64 + mi * 16 + l16;
#pragma unroll
            for (int e = 0; e < 16; ++e) {
                int kk = ((e < 8) ? 0 : 16) + g * 8 + (e & 7);   // 2x 16B runs
                af[mi][e] = As[m][kk];
            }
        }
        v16bf bfr[2];
#pragma unroll
        for (int ni = 0; ni < 2; ++ni) {
            int n = wn * 32 + ni * 16 + l16;
#pragma unroll
            for (int e = 0; e < 16; ++e)
                bfr[ni][e] = Bst[n][((e < 8) ? 0 : 16) + g * 8 + (e & 7)];   // 2x 16B runs
        }
#pragma unroll
        for (int mi = 0; mi < 4; ++mi)
#pragma unroll
            for (int ni = 0; ni < 2; ++ni)
                acc[mi][ni] = __builtin_amdgcn_wmma_f32_16x16x32_bf16(
                    false, af[mi], false, bfr[ni], (short)0, acc[mi][ni], false, false);
        __syncthreads();
    }

#pragma unroll
    for (int mi = 0; mi < 4; ++mi)
#pragma unroll
        for (int ni = 0; ni < 2; ++ni) {
            int n = tile_n + wn * 32 + ni * 16 + l16;
            float bv = (bias != nullptr) ? bias[n] : 0.0f;
#pragma unroll
            for (int r = 0; r < 8; ++r) {
                int m = tile_m + wm * 64 + mi * 16 + g * 8 + r;
                float v = acc[mi][ni][r] + bv;
                if (resid != nullptr) v += resid[(size_t)m * N + n];
                if (relu) v = v > 0.0f ? v : 0.0f;
                if (Cf != nullptr) Cf[(size_t)m * N + n] = v;
                if (Cb != nullptr) Cb[(size_t)m * N + n] = (bf16_t)v;
            }
        }
}

// ---------------------------------------------------------------------------
// Fused causal flash attention, one (b, h, 64-row q-block) per 128-thread
// (4-wave) block. Q.K^T and P.V via WMMA bf16; online softmax in VGPRs.
// All LDS tiles fragment-contiguous, rows padded 64->72 (144B = 36 dwords,
// conflict-free across 16 lanes, 16B aligned).
//   Qs[m][d], Ks[key][d] (natural), Vt[d][key] (transposed), Ps[m][key].
// ---------------------------------------------------------------------------
#define AT_P 72

__global__ __launch_bounds__(128)
void attn_kernel(const bf16_t* __restrict__ q, const bf16_t* __restrict__ k,
                 const bf16_t* __restrict__ v, bf16_t* __restrict__ o,
                 int T, int D, float scale) {
    __shared__ bf16_t Qs[64][AT_P];
    __shared__ bf16_t Ks[64][AT_P];
    __shared__ bf16_t Vt[64][AT_P];
    __shared__ bf16_t Ps[64][AT_P];

    const int tid  = threadIdx.x;
    const int lane = tid & 31;
    const int wave = tid >> 5;     // 0..3, owns rows wave*16..wave*16+15
    const int g    = lane >> 4;
    const int l16  = lane & 15;
    const int qb   = blockIdx.x;
    const int h    = blockIdx.y;
    const int b    = blockIdx.z;
    const int qb0  = qb * 64;

    // load Q block (natural layout, b128 copies)
    for (int i = tid; i < 64 * 64 / 8; i += 128) {
        int m = (i * 8) / 64, c = (i * 8) % 64;
        *(uint4*)&Qs[m][c] =
            *(const uint4*)&q[((size_t)(b * T + qb0 + m)) * D + h * 64 + c];
    }
    __syncthreads();

    // pre-build Q A-fragments (K = 64 -> 2 fragments of K=32)
    v16bf qa[2];
#pragma unroll
    for (int ks = 0; ks < 2; ++ks)
#pragma unroll
        for (int e = 0; e < 16; ++e) {
            int kk = ks * 32 + ((e < 8) ? 0 : 16) + g * 8 + (e & 7);
            qa[ks][e] = Qs[wave * 16 + l16][kk];
        }

    v8f accO[4];
#pragma unroll
    for (int ni = 0; ni < 4; ++ni) accO[ni] = v8f_zero();
    float mrow[8], lrow[8];
#pragma unroll
    for (int r = 0; r < 8; ++r) { mrow[r] = NEG_BIG; lrow[r] = 0.0f; }

    for (int kb = 0; kb <= qb; ++kb) {
        __syncthreads();   // protect Ks/Vt reuse
        // K block: natural [key][d] b128 copy
        for (int i = tid; i < 64 * 64 / 8; i += 128) {
            int n = (i * 8) / 64, c = (i * 8) % 64;
            *(uint4*)&Ks[n][c] =
                *(const uint4*)&k[((size_t)(b * T + kb * 64 + n)) * D + h * 64 + c];
        }
        // V block: b128 read along d, transposed store Vt[d][key]
        for (int i = tid; i < 64 * 64 / 8; i += 128) {
            int key = (i * 8) / 64, d = (i * 8) % 64;
            uint4 w =
                *(const uint4*)&v[((size_t)(b * T + kb * 64 + key)) * D + h * 64 + d];
            const bf16_t* e = (const bf16_t*)&w;
#pragma unroll
            for (int j = 0; j < 8; ++j) Vt[d + j][key] = e[j];
        }
        __syncthreads();

        // S = Q @ K^T  (16x64 per wave); B[kk=d][n=key] = Ks[key][d]
        v8f sacc[4];
#pragma unroll
        for (int ni = 0; ni < 4; ++ni) sacc[ni] = v8f_zero();
#pragma unroll
        for (int ks = 0; ks < 2; ++ks)
#pragma unroll
            for (int ni = 0; ni < 4; ++ni) {
                v16bf bfrag;
#pragma unroll
                for (int e = 0; e < 16; ++e)
                    bfrag[e] = Ks[ni * 16 + l16][ks * 32 + ((e < 8) ? 0 : 16) + g * 8 + (e & 7)];
                sacc[ni] = __builtin_amdgcn_wmma_f32_16x16x32_bf16(
                    false, qa[ks], false, bfrag, (short)0, sacc[ni], false, false);
            }

        // scale + causal mask
        float sv[4][8];
#pragma unroll
        for (int ni = 0; ni < 4; ++ni) {
            int ksn = kb * 64 + ni * 16 + l16;
#pragma unroll
            for (int r = 0; r < 8; ++r) {
                int tq = qb0 + wave * 16 + g * 8 + r;
                float x = sacc[ni][r] * scale;
                sv[ni][r] = (ksn <= tq) ? x : NEG_BIG;
            }
        }

        // online softmax: row max / rescale / exp / row sum
#pragma unroll
        for (int r = 0; r < 8; ++r) {
            float mx = fmaxf(fmaxf(sv[0][r], sv[1][r]), fmaxf(sv[2][r], sv[3][r]));
            mx = fmaxf(mx, __shfl_xor(mx, 1, 32));
            mx = fmaxf(mx, __shfl_xor(mx, 2, 32));
            mx = fmaxf(mx, __shfl_xor(mx, 4, 32));
            mx = fmaxf(mx, __shfl_xor(mx, 8, 32));
            float mn = fmaxf(mrow[r], mx);
            float corr = __expf(mrow[r] - mn);
            mrow[r] = mn;
            lrow[r] *= corr;
#pragma unroll
            for (int ni = 0; ni < 4; ++ni) accO[ni][r] *= corr;
        }
#pragma unroll
        for (int r = 0; r < 8; ++r) {
            float rs = 0.0f;
#pragma unroll
            for (int ni = 0; ni < 4; ++ni) {
                float p = __expf(sv[ni][r] - mrow[r]);
                rs += p;
                Ps[wave * 16 + g * 8 + r][ni * 16 + l16] = (bf16_t)p;
            }
            rs += __shfl_xor(rs, 1, 32);
            rs += __shfl_xor(rs, 2, 32);
            rs += __shfl_xor(rs, 4, 32);
            rs += __shfl_xor(rs, 8, 32);
            lrow[r] += rs;
        }
        __syncthreads();

        // O += P @ V ; B[kk=key][n=d] = Vt[d][key]
#pragma unroll
        for (int ks = 0; ks < 2; ++ks) {
            v16bf pa;
#pragma unroll
            for (int e = 0; e < 16; ++e) {
                int kk = ks * 32 + ((e < 8) ? 0 : 16) + g * 8 + (e & 7);
                pa[e] = Ps[wave * 16 + l16][kk];
            }
#pragma unroll
            for (int ni = 0; ni < 4; ++ni) {
                v16bf vfrag;
#pragma unroll
                for (int e = 0; e < 16; ++e)
                    vfrag[e] = Vt[ni * 16 + l16][ks * 32 + ((e < 8) ? 0 : 16) + g * 8 + (e & 7)];
                accO[ni] = __builtin_amdgcn_wmma_f32_16x16x32_bf16(
                    false, pa, false, vfrag, (short)0, accO[ni], false, false);
            }
        }
    }

    // normalize and write
#pragma unroll
    for (int ni = 0; ni < 4; ++ni)
#pragma unroll
        for (int r = 0; r < 8; ++r) {
            int m = wave * 16 + g * 8 + r;
            float val = accO[ni][r] / lrow[r];
            o[((size_t)(b * T + qb0 + m)) * D + h * 64 + ni * 16 + l16] = (bf16_t)val;
        }
}

// ---------------------------------------------------------------------------
// Per-row NLL term: lse(row) - logits[row, target[row]]
// ---------------------------------------------------------------------------
__global__ __launch_bounds__(256)
void row_nll_kernel(const float* __restrict__ logits, const int* __restrict__ targets,
                    float* __restrict__ terms, int V) {
    int row = blockIdx.x;
    int tid = threadIdx.x;
    const float* lr = logits + (size_t)row * V;
    __shared__ float red[256];

    float mx = NEG_BIG;
    for (int c = tid; c < V; c += 256) mx = fmaxf(mx, lr[c]);
    red[tid] = mx; __syncthreads();
    for (int st = 128; st > 0; st >>= 1) {
        if (tid < st) red[tid] = fmaxf(red[tid], red[tid + st]);
        __syncthreads();
    }
    float M = red[0];
    __syncthreads();

    float s = 0.0f;
    for (int c = tid; c < V; c += 256) s += __expf(lr[c] - M);
    red[tid] = s; __syncthreads();
    for (int st = 128; st > 0; st >>= 1) {
        if (tid < st) red[tid] += red[tid + st];
        __syncthreads();
    }
    if (tid == 0) terms[row] = M + logf(red[0]) - lr[targets[row]];
}

__global__ __launch_bounds__(256)
void loss_reduce_kernel(const float* __restrict__ terms, float* __restrict__ loss_out, int n) {
    __shared__ float red[256];
    int tid = threadIdx.x;
    float s = 0.0f;
    for (int i = tid; i < n; i += 256) s += terms[i];
    red[tid] = s; __syncthreads();
    for (int st = 128; st > 0; st >>= 1) {
        if (tid < st) red[tid] += red[tid + st];
        __syncthreads();
    }
    if (tid == 0) *loss_out = red[0] / (float)n;
}

// ---------------------------------------------------------------------------
// Host orchestration
// ---------------------------------------------------------------------------
extern "C" void kernel_launch(void* const* d_in, const int* in_sizes, int n_in,
                              void* d_out, int out_size, void* d_ws, size_t ws_size,
                              hipStream_t stream) {
    (void)in_sizes; (void)n_in; (void)out_size; (void)ws_size;
    const int Bz = 4, T = 1024, D = 1024, Hh = 16, HS = 64, Db = 64;
    const int Vv = 32000, Lc = 6, DFF = 4 * D;
    const int M = Bz * T;

    const int*   idx     = (const int*)d_in[0];
    const int*   ibx     = (const int*)d_in[1];
    const int*   targets = (const int*)d_in[2];
    const float* tok     = (const float*)d_in[3];
    const float* pos     = (const float*)d_in[4];
    const float* book    = (const float*)d_in[5];
    const float* Wq      = (const float*)d_in[6];
    const float* Wk      = (const float*)d_in[7];
    const float* Wv      = (const float*)d_in[8];
    const float* Wo      = (const float*)d_in[9];
    const float* bo      = (const float*)d_in[10];
    const float* ln1_s   = (const float*)d_in[11];
    const float* ln1_b   = (const float*)d_in[12];
    const float* ln2_s   = (const float*)d_in[13];
    const float* ln2_b   = (const float*)d_in[14];
    const float* W1      = (const float*)d_in[15];
    const float* b1      = (const float*)d_in[16];
    const float* W2      = (const float*)d_in[17];
    const float* b2      = (const float*)d_in[18];
    const float* lnf_s   = (const float*)d_in[19];
    const float* lnf_b   = (const float*)d_in[20];
    const float* Wh      = (const float*)d_in[21];
    const float* bh      = (const float*)d_in[22];

    float* logits = (float*)d_out;
    float* loss   = logits + (size_t)M * Vv;

    // workspace carve-out (~321 MB)
    char* wsp = (char*)d_ws;
    auto carve = [&](size_t bytes) {
        char* p = wsp;
        wsp += (bytes + 255) & ~(size_t)255;
        return p;
    };
    bf16_t* wq_b  = (bf16_t*)carve((size_t)Lc * D * D * 2);
    bf16_t* wk_b  = (bf16_t*)carve((size_t)Lc * D * D * 2);
    bf16_t* wv_b  = (bf16_t*)carve((size_t)Lc * D * D * 2);
    bf16_t* wo_b  = (bf16_t*)carve((size_t)Lc * D * D * 2);
    bf16_t* w1_b  = (bf16_t*)carve((size_t)Lc * (D + Db) * DFF * 2);
    bf16_t* w2_b  = (bf16_t*)carve((size_t)Lc * DFF * D * 2);
    bf16_t* wh_b  = (bf16_t*)carve((size_t)D * Vv * 2);
    float*  x     = (float*)carve((size_t)M * D * 4);
    bf16_t* h_b   = (bf16_t*)carve((size_t)M * D * 2);
    bf16_t* q_b   = (bf16_t*)carve((size_t)M * D * 2);
    bf16_t* k_b   = (bf16_t*)carve((size_t)M * D * 2);
    bf16_t* v_b   = (bf16_t*)carve((size_t)M * D * 2);
    bf16_t* o_b   = (bf16_t*)carve((size_t)M * D * 2);
    bf16_t* h2_b  = (bf16_t*)carve((size_t)M * (D + Db) * 2);
    bf16_t* ff_b  = (bf16_t*)carve((size_t)M * DFF * 2);
    float*  terms = (float*)carve((size_t)M * 4);

    auto cvt = [&](const float* in, bf16_t* out, size_t n) {
        cvt_bf16_kernel<<<dim3(2048), dim3(256), 0, stream>>>(in, out, n);
    };
    cvt(Wq, wq_b, (size_t)Lc * D * D);
    cvt(Wk, wk_b, (size_t)Lc * D * D);
    cvt(Wv, wv_b, (size_t)Lc * D * D);
    cvt(Wo, wo_b, (size_t)Lc * D * D);
    cvt(W1, w1_b, (size_t)Lc * (D + Db) * DFF);
    cvt(W2, w2_b, (size_t)Lc * DFF * D);
    cvt(Wh, wh_b, (size_t)D * Vv);

    embed_kernel<<<dim3(M), dim3(256), 0, stream>>>(idx, tok, pos, x, T, D);

    auto gemm = [&](const bf16_t* A, const bf16_t* Bw, const float* bias,
                    const float* res, float* Cf, bf16_t* Cb,
                    int Mm, int Nn, int Kk, int relu) {
        gemm_bf16_kernel<<<dim3(Nn / 128, Mm / 128), dim3(256), 0, stream>>>(
            A, Bw, bias, res, Cf, Cb, Mm, Nn, Kk, relu);
    };
    auto ln = [&](const float* xin, const float* gam, const float* bet,
                  bf16_t* out, int stride, const float* bk) {
        layernorm_kernel<<<dim3(M), dim3(256), 0, stream>>>(
            xin, gam, bet, out, D, stride, bk, ibx, Db, T);
    };

    const float scale = 1.0f / sqrtf((float)HS);
    for (int l = 0; l < Lc; ++l) {
        size_t wdd = (size_t)l * D * D;
        // h = LN1(x)  (bf16)
        ln(x, ln1_s + (size_t)l * D, ln1_b + (size_t)l * D, h_b, D, nullptr);
        // q,k,v = h @ W{q,k,v}
        gemm(h_b, wq_b + wdd, nullptr, nullptr, nullptr, q_b, M, D, D, 0);
        gemm(h_b, wk_b + wdd, nullptr, nullptr, nullptr, k_b, M, D, D, 0);
        gemm(h_b, wv_b + wdd, nullptr, nullptr, nullptr, v_b, M, D, D, 0);
        // o = causal softmax(q k^T / sqrt(HS)) v
        attn_kernel<<<dim3(T / 64, Hh, Bz), dim3(128), 0, stream>>>(
            q_b, k_b, v_b, o_b, T, D, scale);
        // x = x + o @ Wo + bo
        gemm(o_b, wo_b + wdd, bo + (size_t)l * D, x, x, nullptr, M, D, D, 0);
        // h2 = concat(LN2(x), book)  (bf16, stride D+Db)
        ln(x, ln2_s + (size_t)l * D, ln2_b + (size_t)l * D, h2_b, D + Db, book);
        // ff = relu(h2 @ W1 + b1)  (bf16)
        gemm(h2_b, w1_b + (size_t)l * (D + Db) * DFF, b1 + (size_t)l * DFF,
             nullptr, nullptr, ff_b, M, DFF, D + Db, 1);
        // x = x + ff @ W2 + b2
        gemm(ff_b, w2_b + (size_t)l * DFF * D, b2 + (size_t)l * D, x, x, nullptr,
             M, D, DFF, 0);
    }

    // final LN -> bf16, LM head -> logits (f32, straight into d_out)
    ln(x, lnf_s, lnf_b, h_b, D, nullptr);
    gemm(h_b, wh_b, bh, nullptr, logits, nullptr, M, Vv, D, 0);

    // loss = mean(lse - logit[target]) (deterministic tree reductions)
    row_nll_kernel<<<dim3(M), dim3(256), 0, stream>>>(logits, targets, terms, Vv);
    loss_reduce_kernel<<<dim3(1), dim3(256), 0, stream>>>(terms, loss, M);
}
